// AdaptiveProteinBlock_68204080660482
// MI455X (gfx1250) — matrix operand order repaired and checked by reading, not compiled, not run on hardware
//
#include <hip/hip_runtime.h>

#define NROWS 8192
#define DIN   512
#define DATT  64
#define TOPK  16
#define LN_EPS 1e-5f
#define CHUNK 512     // attention column chunk held in LDS (512*16*4B = 32KB)
#define LDPAD 8       // 8 bf16 = 4 DWORDs LDS row padding (bank-conflict-free B frags)

typedef __attribute__((ext_vector_type(16))) __bf16 bf16x16;
typedef __attribute__((ext_vector_type(8)))  __bf16 bf16x8;
typedef __attribute__((ext_vector_type(8)))  float  f32x8;
typedef __attribute__((ext_vector_type(4)))  unsigned int u32x4;
typedef __attribute__((ext_vector_type(8)))  int i32x8;
typedef __attribute__((ext_vector_type(4)))  int i32x4;

// ---------------------------------------------------------------- WMMA helpers
__device__ __forceinline__ f32x8 wmma_bf16(bf16x16 a, bf16x16 b, f32x8 c) {
  // v_wmma_f32_16x16x32_bf16  (neg_a, A, neg_b, B, c_mod, C, reuse_a, reuse_b)
  return __builtin_amdgcn_wmma_f32_16x16x32_bf16(false, a, false, b, (short)0, c, false, false);
}

// A-matrix fragment, 16x32 bf16, row-major source, leading dim `ld`.
// ISA 7.12.2: lanes 0-15 row M=lane hold K {0..7,16..23};
// lanes 16-31 row M=lane-16 hold K {8..15,24..31}.
__device__ __forceinline__ bf16x16 load_a_frag(const __bf16* src, int ld,
                                               int row0, int kb, int lane) {
  const int half = lane >> 4, m = lane & 15;
  const __bf16* p = src + (size_t)(row0 + m) * ld + kb + half * 8;
  bf16x8 lo = *(const bf16x8*)p;          // K = base+0..7
  bf16x8 hi = *(const bf16x8*)(p + 16);   // K = base+16..23
  bf16x16 a;
#pragma unroll
  for (int i = 0; i < 8; ++i) { a[i] = lo[i]; a[i + 8] = hi[i]; }
  return a;
}

// B-matrix fragment, 32x16 bf16, for C = A*B^T with B row-major [Nc, K]:
// lane n holds column n0+n; lanes 0-15 carry K {0..7,16..23}, lanes 16-31 K {8..15,24..31} (rel. kb).
__device__ __forceinline__ bf16x16 load_b_frag(const __bf16* B, int ld,
                                               int n0, int kb, int lane) {
  const int half = lane >> 4, n = lane & 15;
  const __bf16* p = B + (size_t)(n0 + n) * ld + kb + half * 8;
  bf16x8 lo = *(const bf16x8*)p;
  bf16x8 hi = *(const bf16x8*)(p + 16);
  bf16x16 b;
#pragma unroll
  for (int i = 0; i < 8; ++i) { b[i] = lo[i]; b[i + 8] = hi[i]; }
  return b;
}

// ---------------------------------------------------------------- converts
__global__ __launch_bounds__(256) void cvt_bf16(const float* __restrict__ in,
                                                __bf16* __restrict__ out, int n) {
  int i = blockIdx.x * 256 + threadIdx.x;
  if (i < n) out[i] = (__bf16)in[i];
}

// out[n][k] = in[k][n] for 64x64 (W2 -> W2^T so GEMM computes Q@W2)
__global__ __launch_bounds__(64) void transpose64(const float* __restrict__ in,
                                                  __bf16* __restrict__ out) {
  out[blockIdx.x * 64 + threadIdx.x] = (__bf16)in[threadIdx.x * 64 + blockIdx.x];
}

// ---------------------------------------------------------------- TDM B-panel stage
// Stage B rows [n0, n0+32) x K into LDS (row stride Kd+LDPAD bf16) via the
// Tensor Data Mover: 2D tile, data_size=2B, pad_enable (4 DWORDs per row) so
// B-fragment ds reads land on distinct banks. Issued by one wave, tracked by
// TENSORcnt, published with a workgroup barrier.
__device__ __forceinline__ void tdm_stage_B(const __bf16* B, int Kd, int n0,
                                            __bf16* sB, int tid, int wave) {
#if defined(__HIP_DEVICE_COMPILE__) && __has_builtin(__builtin_amdgcn_tensor_load_to_lds)
  if (wave == 0) {
    const unsigned long long ga = (unsigned long long)(uintptr_t)(B + (size_t)n0 * Kd);
    const unsigned lds = (unsigned)(uintptr_t)(void*)sB;   // addr[31:0] = LDS offset
    const int padint = __builtin_ctz((unsigned)Kd) - 2;    // Kd/2 DWORDs per row
    u32x4 g0;
    g0[0] = 1u;                                            // count=1 (valid), user mode
    g0[1] = lds;                                           // lds_addr  [63:32]
    g0[2] = (unsigned)(ga & 0xFFFFFFFFu);                  // global_addr[95:64]
    g0[3] = (unsigned)((ga >> 32) & 0x1FFFFFFu) | (2u << 30); // addr[56:32] | type=2
    i32x8 g1;
    g1[0] = (int)((1u << 16) |                             // data_size = 2 bytes
                  (1u << 20) |                             // pad_enable
                  ((unsigned)padint << 22) |               // pad_interval = one row
                  (3u << 25));                             // pad_amount = 4 DWORDs
    g1[1] = (int)(((unsigned)Kd & 0xFFFFu) << 16);         // tensor_dim0 lo16
    g1[2] = (int)(((unsigned)Kd >> 16) | (32u << 16));     // tensor_dim0 hi | tensor_dim1=32
    g1[3] = (int)((unsigned)Kd << 16);                     // tile_dim0 = Kd
    g1[4] = 32;                                            // tile_dim1 = 32
    g1[5] = Kd;                                            // tensor_dim0_stride = Kd
    g1[6] = 0; g1[7] = 0;
    i32x4 gz = {0, 0, 0, 0};
#if __clang_major__ >= 23
    i32x8 gz8 = {0, 0, 0, 0, 0, 0, 0, 0};
    __builtin_amdgcn_tensor_load_to_lds(g0, g1, gz, gz, gz8, 0);
#else
    __builtin_amdgcn_tensor_load_to_lds(g0, g1, gz, gz, 0);
#endif
    __builtin_amdgcn_s_wait_tensorcnt(0);
  }
#else
  // cooperative fallback: rows are contiguous in B; copy with padded LDS stride
  const int rowsz = Kd / 8;                   // bf16x8 chunks per row
  for (int i = tid; i < 32 * rowsz; i += 256) {
    const int r = i / rowsz, k = i - r * rowsz;
    *(bf16x8*)(sB + (size_t)r * (Kd + LDPAD) + k * 8) =
        *(const bf16x8*)(B + (size_t)(n0 + r) * Kd + k * 8);
  }
#endif
  __syncthreads();
}

// ---------------------------------------------------------------- GEMM (TDM + 2x2 register blocking)
// C[M,Nc] = A[M,K] * B[Nc,K]^T (+bias). Block = 8 waves = 256 rows x 32 cols;
// B panel staged in LDS once per block; each wave owns a 32x32 macro-tile
// (4 accumulators, 4 WMMAs per 2 A-frag + 2 B-frag loads).
__global__ __launch_bounds__(256) void gemm_bt_bf16(
    const __bf16* __restrict__ A, const __bf16* __restrict__ B,
    float* __restrict__ outF, __bf16* __restrict__ outB,
    const float* __restrict__ bias, int M, int Nc, int Kd, int accumulate) {
  extern __shared__ __bf16 sB[];            // [32][Kd + LDPAD]
  const int lane = threadIdx.x & 31;
  const int wave = threadIdx.x >> 5;
  const int nblk = Nc >> 5;
  const int cb = blockIdx.x % nblk;
  const int rb = blockIdx.x / nblk;
  const int n0 = cb << 5;
  const int row0 = rb * 256 + wave * 32;
  const int ldb = Kd + LDPAD;

  tdm_stage_B(B, Kd, n0, sB, threadIdx.x, wave);

  f32x8 acc[2][2];
#pragma unroll
  for (int i = 0; i < 2; ++i)
#pragma unroll
    for (int j = 0; j < 2; ++j) acc[i][j] = (f32x8){};

  for (int kb = 0; kb < Kd; kb += 32) {
    bf16x16 a0 = load_a_frag(A, Kd, row0, kb, lane);
    bf16x16 a1 = load_a_frag(A, Kd, row0 + 16, kb, lane);
    if (kb + 32 < Kd) {
      __builtin_prefetch(A + (size_t)(row0 + (lane & 15)) * Kd + kb + 32, 0, 3);
      __builtin_prefetch(A + (size_t)(row0 + 16 + (lane & 15)) * Kd + kb + 32, 0, 3);
    }
    bf16x16 b0 = load_b_frag(sB, ldb, 0, kb, lane);   // ds reads, padded stride
    bf16x16 b1 = load_b_frag(sB, ldb, 16, kb, lane);
    acc[0][0] = wmma_bf16(a0, b0, acc[0][0]);
    acc[0][1] = wmma_bf16(a0, b1, acc[0][1]);
    acc[1][0] = wmma_bf16(a1, b0, acc[1][0]);
    acc[1][1] = wmma_bf16(a1, b1, acc[1][1]);
  }

  const int col = lane & 15, half = lane >> 4;
#pragma unroll
  for (int tr = 0; tr < 2; ++tr)
#pragma unroll
    for (int tc = 0; tc < 2; ++tc) {
      const int cc = n0 + tc * 16 + col;
      const float bv = bias ? bias[cc] : 0.f;
#pragma unroll
      for (int v = 0; v < 8; ++v) {         // row = v + 8*half (ISA C layout)
        const size_t off = (size_t)(row0 + tr * 16 + v + 8 * half) * Nc + cc;
        float val = acc[tr][tc][v] + bv;
        if (outF) outF[off] = accumulate ? (outF[off] + val) : val;
        if (outB) outB[off] = (__bf16)val;
      }
    }
}

// ---------------------------------------------------------------- fused S-tile + top-k + softmax
// Block = 16 rows. 8 waves sweep 8192 columns in CHUNK-wide LDS slabs.
// S (256 MB dense) never touches global memory.
__global__ __launch_bounds__(256) void attn_topk(
    const __bf16* __restrict__ Abf,   // (Q@W2) in bf16, [N,64]
    const __bf16* __restrict__ Kbf,   // K in bf16, [N,64]
    float* __restrict__ wgt, int* __restrict__ idxout) {
  __shared__ float Sch[CHUNK * 16];   // [col][row], 32 KB
  const int lane = threadIdx.x & 31;
  const int wave = threadIdx.x >> 5;
  const int row0 = blockIdx.x << 4;

  // A rows fixed for the whole block: K=64 -> two resident fragments
  bf16x16 a0 = load_a_frag(Abf, DATT, row0, 0, lane);
  bf16x16 a1 = load_a_frag(Abf, DATT, row0, 32, lane);

  float tv[TOPK]; int ti[TOPK];
#pragma unroll
  for (int t = 0; t < TOPK; ++t) { tv[t] = -INFINITY; ti[t] = 0; }
  float minv = -INFINITY; int minp = 0;

  for (int ch = 0; ch < NROWS / CHUNK; ++ch) {
    const int colbase = ch * CHUNK;
#pragma unroll
    for (int t4 = 0; t4 < CHUNK / (8 * 16); ++t4) {   // 4 S-tiles per wave
      const int n0 = colbase + wave * (CHUNK / 8) + t4 * 16;
      bf16x16 b0 = load_b_frag(Kbf, DATT, n0, 0, lane);
      bf16x16 b1 = load_b_frag(Kbf, DATT, n0, 32, lane);
      f32x8 c = {};
      c = wmma_bf16(a0, b0, c);
      c = wmma_bf16(a1, b1, c);
      const int lcol = (n0 - colbase) + (lane & 15);
      const int half = lane >> 4;
#pragma unroll
      for (int v = 0; v < 8; ++v)
        Sch[lcol * 16 + v + 8 * half] = c[v];   // TAU = 1.0
    }
    __syncthreads();
    if (wave == 0 && lane < 16) {               // lane = row owner; no WMMA here
      for (int j = 0; j < CHUNK; ++j) {
        float s = Sch[j * 16 + lane];
        if (s > minv) {                          // replace current min, unrolled
#pragma unroll
          for (int t = 0; t < TOPK; ++t)
            if (t == minp) { tv[t] = s; ti[t] = colbase + j; }
          minv = tv[0]; minp = 0;
#pragma unroll
          for (int t = 1; t < TOPK; ++t)
            if (tv[t] < minv) { minv = tv[t]; minp = t; }
        }
      }
    }
    __syncthreads();
  }

  if (wave == 0 && lane < 16) {                  // softmax over the 16 survivors
    float mx = tv[0];
#pragma unroll
    for (int t = 1; t < TOPK; ++t) mx = fmaxf(mx, tv[t]);
    float e[TOPK]; float sum = 0.f;
#pragma unroll
    for (int t = 0; t < TOPK; ++t) { e[t] = __expf(tv[t] - mx); sum += e[t]; }
    const float inv = 1.f / sum;
    const int r = row0 + lane;
#pragma unroll
    for (int t = 0; t < TOPK; ++t) {
      wgt[r * TOPK + t] = e[t] * inv;
      idxout[r * TOPK + t] = ti[t];
    }
  }
}

// ---------------------------------------------------------------- P @ H (row gather, k=16)
__global__ __launch_bounds__(256) void gather_rows(
    const float* __restrict__ Hin, const float* __restrict__ wgt,
    const int* __restrict__ idx, float* __restrict__ HoutF,
    __bf16* __restrict__ HoutB) {
  const int row = blockIdx.x, tid = threadIdx.x;
  __shared__ float wl[TOPK]; __shared__ int il[TOPK];
  if (tid < TOPK) { wl[tid] = wgt[row * TOPK + tid]; il[tid] = idx[row * TOPK + tid]; }
  __syncthreads();
#pragma unroll
  for (int c = tid; c < DIN; c += 256) {
    float acc = 0.f;
#pragma unroll
    for (int t = 0; t < TOPK; ++t)
      acc += wl[t] * Hin[(size_t)il[t] * DIN + c];
    if (HoutF) HoutF[(size_t)row * DIN + c] = acc;
    HoutB[(size_t)row * DIN + c] = (__bf16)acc;
  }
}

// ---------------------------------------------------------------- residual + LayerNorm
__global__ __launch_bounds__(256) void ln_kernel(
    const float* __restrict__ X, const float* __restrict__ Z,
    const float* __restrict__ gamma, const float* __restrict__ beta,
    float* __restrict__ out) {
  const int row = blockIdx.x, tid = threadIdx.x;
  const float* xr = X + (size_t)row * DIN;
  const float* zr = Z + (size_t)row * DIN;
  float y0 = xr[tid] + zr[tid];
  float y1 = xr[tid + 256] + zr[tid + 256];
  __shared__ float sh[256];
  sh[tid] = y0 + y1;
  __syncthreads();
  for (int s = 128; s > 0; s >>= 1) { if (tid < s) sh[tid] += sh[tid + s]; __syncthreads(); }
  const float mu = sh[0] * (1.f / DIN);
  __syncthreads();
  const float d0 = y0 - mu, d1 = y1 - mu;
  sh[tid] = d0 * d0 + d1 * d1;
  __syncthreads();
  for (int s = 128; s > 0; s >>= 1) { if (tid < s) sh[tid] += sh[tid + s]; __syncthreads(); }
  const float rstd = rsqrtf(sh[0] * (1.f / DIN) + LN_EPS);
  out[(size_t)row * DIN + tid]       = d0 * rstd * gamma[tid] + beta[tid];
  out[(size_t)row * DIN + tid + 256] = d1 * rstd * gamma[tid + 256] + beta[tid + 256];
}

// ---------------------------------------------------------------- launcher
extern "C" void kernel_launch(void* const* d_in, const int* in_sizes, int n_in,
                              void* d_out, int out_size, void* d_ws, size_t ws_size,
                              hipStream_t stream) {
  const float* X     = (const float*)d_in[0];   // [8192,512]
  const float* W1    = (const float*)d_in[1];   // [64,512]
  const float* W2    = (const float*)d_in[2];   // [64,64]
  const float* W3    = (const float*)d_in[3];   // [64,512]
  const float* mixW  = (const float*)d_in[4];   // [2,512,512]
  const float* mixB  = (const float*)d_in[5];   // [2,512]
  const float* gamma = (const float*)d_in[6];
  const float* beta  = (const float*)d_in[7];
  float* out = (float*)d_out;

  char* ws = (char*)d_ws;
  size_t o = 0;
  auto take = [&](size_t bytes) -> char* {
    char* p = ws + o; o += (bytes + 255) & ~(size_t)255; return p;
  };
  __bf16* Xbf   = (__bf16*)take((size_t)NROWS * DIN * 2);
  __bf16* W1bf  = (__bf16*)take((size_t)DATT * DIN * 2);
  __bf16* W3bf  = (__bf16*)take((size_t)DATT * DIN * 2);
  __bf16* W2T   = (__bf16*)take((size_t)DATT * DATT * 2);
  __bf16* mixWb = (__bf16*)take((size_t)2 * DIN * DIN * 2);
  __bf16* Qbf   = (__bf16*)take((size_t)NROWS * DATT * 2);
  __bf16* Kbf   = (__bf16*)take((size_t)NROWS * DATT * 2);
  __bf16* Abf   = (__bf16*)take((size_t)NROWS * DATT * 2);
  int*    idx   = (int*)  take((size_t)NROWS * TOPK * 4);
  float*  wgt   = (float*)take((size_t)NROWS * TOPK * 4);
  float*  H1f   = (float*)take((size_t)NROWS * DIN * 4);
  __bf16* H1bf  = (__bf16*)take((size_t)NROWS * DIN * 2);
  __bf16* H2bf  = (__bf16*)take((size_t)NROWS * DIN * 2);
  float*  Zf    = (float*)take((size_t)NROWS * DIN * 4);
  (void)ws_size; (void)n_in; (void)in_sizes; (void)out_size;

  const int smemK512 = 32 * (DIN + LDPAD) * 2;   // B panel, K=512
  const int smemK64  = 32 * (DATT + LDPAD) * 2;  // B panel, K=64

  // 1) precision conversion
  cvt_bf16<<<(NROWS * DIN + 255) / 256, 256, 0, stream>>>(X, Xbf, NROWS * DIN);
  cvt_bf16<<<(DATT * DIN + 255) / 256, 256, 0, stream>>>(W1, W1bf, DATT * DIN);
  cvt_bf16<<<(DATT * DIN + 255) / 256, 256, 0, stream>>>(W3, W3bf, DATT * DIN);
  cvt_bf16<<<(2 * DIN * DIN + 255) / 256, 256, 0, stream>>>(mixW, mixWb, 2 * DIN * DIN);
  transpose64<<<64, 64, 0, stream>>>(W2, W2T);

  // 2) projections: Q = X @ W1^T, K = X @ W3^T   (grid = (8192/256)*(64/32) = 64)
  gemm_bt_bf16<<<64, 256, smemK512, stream>>>(Xbf, W1bf, nullptr, Qbf, nullptr, NROWS, DATT, DIN, 0);
  gemm_bt_bf16<<<64, 256, smemK512, stream>>>(Xbf, W3bf, nullptr, Kbf, nullptr, NROWS, DATT, DIN, 0);
  // A = Q @ W2  (== Q @ (W2^T)^T)
  gemm_bt_bf16<<<64, 256, smemK64, stream>>>(Qbf, W2T, nullptr, Abf, nullptr, NROWS, DATT, DATT, 0);

  // 3) fused S-tiles + top-16 + softmax (S never materialized)
  attn_topk<<<NROWS / 16, 256, 0, stream>>>(Abf, Kbf, wgt, idx);

  // 4) step 0: H1 = P @ X ; Z = H1 @ mixW0^T + mixB0   (grid = 32*16 = 512)
  gather_rows<<<NROWS, 256, 0, stream>>>(X, wgt, idx, H1f, H1bf);
  gemm_bt_bf16<<<512, 256, smemK512, stream>>>(H1bf, mixWb, Zf, nullptr, mixB, NROWS, DIN, DIN, 0);

  // 5) step 1: H2 = P @ H1 ; Z += H2 @ mixW1^T + mixB1
  gather_rows<<<NROWS, 256, 0, stream>>>(H1f, wgt, idx, nullptr, H2bf);
  gemm_bt_bf16<<<512, 256, smemK512, stream>>>(H2bf, mixWb + (size_t)DIN * DIN, Zf, nullptr,
                                               mixB + DIN, NROWS, DIN, DIN, 1);

  // 6) out = LayerNorm(X + Z) * gamma + beta
  ln_kernel<<<NROWS, 256, 0, stream>>>(X, Zf, gamma, beta, out);
}
